// EntityEncoder_84997402788674
// MI455X (gfx1250) — compile-verified, matrix operand order checked by reading, not run on hardware
//
#include <hip/hip_runtime.h>

typedef __attribute__((ext_vector_type(16))) _Float16 v16h;
typedef __attribute__((ext_vector_type(8)))  _Float16 v8h;
typedef __attribute__((ext_vector_type(8)))  float    v8f;
typedef __attribute__((ext_vector_type(4)))  float    v4f;
typedef __attribute__((ext_vector_type(4)))  _Float16 v4h;
typedef __attribute__((ext_vector_type(4)))  unsigned int u32x4;
typedef __attribute__((ext_vector_type(4)))  int      i32x4;
typedef __attribute__((ext_vector_type(8)))  int      i32x8;

#define HEADS 8
#define HDIM  128
#define ADIM  1024
#define SEQ   1024
#define BATCH 8
#define E0    256
#define INV_SQRT_D 0.08838834764831845f   // 1/sqrt(128)

#if defined(__gfx1250__) && defined(__has_builtin)
#if __has_builtin(__builtin_amdgcn_tensor_load_to_lds)
#define USE_TDM 1
#endif
#endif

// ---------------------------------------------------------------------------
// A/B fragment loader for v_wmma_f32_16x16x32_f16 (ISA 7.12.2 layout):
//   lanes 0-15  : row = lane,    K in {0..7} then {16..23}
//   lanes 16-31 : row = lane-16, K in {8..15} then {24..31}
// ---------------------------------------------------------------------------
static __device__ inline v16h frag_ld(const _Float16* p, int ld) {
  int lane = threadIdx.x & 31;
  int r  = lane & 15;
  int ko = (lane < 16) ? 0 : 8;
  const _Float16* q = p + r * ld + ko;
  v8h lo = *(const v8h*)(q);
  v8h hi = *(const v8h*)(q + 16);
  return __builtin_shufflevector(lo, hi, 0,1,2,3,4,5,6,7,8,9,10,11,12,13,14,15);
}

#if defined(USE_TDM)
// Issue a TDM load of a 128x32 f16 tile of row-major A[M,K] at (bm, k0) into
// LDS byte offset lds_off.  D# group0/group1 per ISA 08_async_tensor §8.
static __device__ inline void tdm_issue(const _Float16* A, int M, int K,
                                        int bm, int k0, unsigned lds_off) {
  unsigned long long ga =
      (unsigned long long)(uintptr_t)(A + (size_t)bm * K + k0);
  u32x4 g0;
  g0[0] = 1u;                                   // count=1, user mode
  g0[1] = lds_off;                              // lds_addr (bytes)
  g0[2] = (unsigned)(ga & 0xFFFFFFFFu);         // global_addr[31:0]
  g0[3] = (unsigned)((ga >> 32) & 0x01FFFFFFu)  // global_addr[56:32]
          | 0x80000000u;                        // type = 2 ("image")
  i32x8 g1;
  g1[0] = (int)(1u << 16);                      // data_size = 1 (2 bytes)
  g1[1] = (int)(((unsigned)K & 0xFFFFu) << 16); // tensor_dim0[15:0]
  g1[2] = (int)((((unsigned)M & 0xFFFFu) << 16) // tensor_dim1[15:0]
                | ((unsigned)K >> 16));         // tensor_dim0[31:16]
  g1[3] = (int)((32u << 16)                     // tile_dim0 = 32
                | ((unsigned)M >> 16));         // tensor_dim1[31:16]
  g1[4] = 128;                                  // tile_dim1=128, tile_dim2=0
  g1[5] = (int)(unsigned)K;                     // tensor_dim0_stride[31:0]
  g1[6] = 0;
  g1[7] = 0;
  i32x4 z4 = {0, 0, 0, 0};
#if __clang_major__ >= 23
  i32x8 z8 = {0, 0, 0, 0, 0, 0, 0, 0};
  __builtin_amdgcn_tensor_load_to_lds(g0, g1, z4, z4, z8, 0);
#else
  __builtin_amdgcn_tensor_load_to_lds(g0, g1, z4, z4, 0);
#endif
}
#endif

// ---------------------------------------------------------------------------
// f32 -> f16 conversion (vectorized, grid-stride)
// ---------------------------------------------------------------------------
__global__ void cvt_f32_to_f16(const float* __restrict__ src,
                               _Float16* __restrict__ dst, long long n) {
  long long i = ((long long)blockIdx.x * blockDim.x + threadIdx.x) * 4;
  long long stride = (long long)gridDim.x * blockDim.x * 4;
  for (; i < n; i += stride) {
    v4f v = *(const v4f*)(src + i);
    v4h h;
    h[0] = (_Float16)v[0]; h[1] = (_Float16)v[1];
    h[2] = (_Float16)v[2]; h[3] = (_Float16)v[3];
    *(v4h*)(dst + i) = h;
  }
}

// ---------------------------------------------------------------------------
// C[M,N] = ReLU(A[M,K] @ W[K,N] + bias)   (A,W f16 row-major; f32 accumulate)
// Block: 256 threads = 8 waves; block tile 128(M) x 64(N); wave tile 32x32.
// Double-buffered software pipeline:
//   - A tiles (128x32) DMA'd by the Tensor Data Mover (wave 0 issues, waits
//     TENSORcnt) one iteration ahead,
//   - W tiles global->regs one iteration ahead, committed transposed to LDS
//     after the current tile's WMMAs are issued,
// so the TDM copy and the W vmem loads fly during the WMMA burst.
// ---------------------------------------------------------------------------
__global__ __launch_bounds__(256) void gemm_relu(
    const _Float16* __restrict__ A, const _Float16* __restrict__ W,
    const float* __restrict__ bias, _Float16* __restrict__ C16,
    float* __restrict__ C32, int M, int N, int K) {
  __shared__ _Float16 As[2][128 * 32];  // row stride 32 halfs (64B)
  __shared__ _Float16 Wt[2][64 * 40];   // 64 cols x 32 K, transposed

  int tid  = threadIdx.x;
  int lane = tid & 31;
  int wave = tid >> 5;
  int wr = wave & 3;        // wave's M sub-tile (0..3) -> 32 rows
  int wc = wave >> 2;       // wave's N sub-tile (0..1) -> 32 cols
  int bm = blockIdx.y * 128;
  int bn = blockIdx.x * 64;

  int kk  = tid & 31;           // W staging coords
  int nn0 = (tid >> 5) * 8;

  auto load_W = [&](int step) {
    return *(const v8h*)(W + (size_t)(step * 32 + kk) * N + bn + nn0);
  };
  auto store_W = [&](v8h wv, int buf) {
#pragma unroll
    for (int e = 0; e < 8; ++e) Wt[buf][(nn0 + e) * 40 + kk] = wv[e];
  };
#if !defined(USE_TDM)
  auto stage_A = [&](int step, int buf) {
    int row = tid >> 1;
    int col = (tid & 1) * 16;
    const _Float16* ap = A + (size_t)(bm + row) * K + step * 32 + col;
    *(v8h*)(&As[buf][row * 32 + col])     = *(const v8h*)(ap);
    *(v8h*)(&As[buf][row * 32 + col + 8]) = *(const v8h*)(ap + 8);
  };
#endif

  v8f acc00 = {}, acc01 = {}, acc10 = {}, acc11 = {};
  const int steps = K >> 5;

  // ---- prologue: stage tile 0 ----
#if defined(USE_TDM)
  unsigned asbase = (unsigned)(unsigned long long)(uintptr_t)(&As[0][0]);
  if (wave == 0) tdm_issue(A, M, K, bm, 0, asbase);
#else
  stage_A(0, 0);
#endif
  v8h wreg = load_W(0);
  store_W(wreg, 0);
#if defined(USE_TDM)
  if (wave == 0) __builtin_amdgcn_s_wait_tensorcnt(0);
#endif
  __syncthreads();

  for (int i = 0; i < steps; ++i) {
    int cur = i & 1;
    int nxt = cur ^ 1;
    bool more = (i + 1) < steps;

    // ---- prefetch tile i+1 (TDM for A, global->regs for W) ----
    if (more) {
#if defined(USE_TDM)
      if (wave == 0)
        tdm_issue(A, M, K, bm, (i + 1) * 32,
                  asbase + (unsigned)nxt * (128 * 32 * 2));
#else
      stage_A(i + 1, nxt);
#endif
      wreg = load_W(i + 1);
    }

    // ---- compute on tile i ----
    v16h af0 = frag_ld(&As[cur][(wr * 32) * 32], 32);
    v16h af1 = frag_ld(&As[cur][(wr * 32 + 16) * 32], 32);
    v16h bf0 = frag_ld(&Wt[cur][(wc * 32) * 40], 40);
    v16h bf1 = frag_ld(&Wt[cur][(wc * 32 + 16) * 40], 40);
    acc00 = __builtin_amdgcn_wmma_f32_16x16x32_f16(false, af0, false, bf0,
                                                   (short)0, acc00, false, false);
    acc01 = __builtin_amdgcn_wmma_f32_16x16x32_f16(false, af0, false, bf1,
                                                   (short)0, acc01, false, false);
    acc10 = __builtin_amdgcn_wmma_f32_16x16x32_f16(false, af1, false, bf0,
                                                   (short)0, acc10, false, false);
    acc11 = __builtin_amdgcn_wmma_f32_16x16x32_f16(false, af1, false, bf1,
                                                   (short)0, acc11, false, false);

    // ---- commit prefetched W, retire the in-flight TDM, swap buffers ----
    if (more) {
      store_W(wreg, nxt);
#if defined(USE_TDM)
      if (wave == 0) __builtin_amdgcn_s_wait_tensorcnt(0);
#endif
      __syncthreads();
    }
  }

  // epilogue: bias + ReLU.  C/D layout: VGPR j -> row j (lanes 0-15) / j+8
  // (lanes 16-31), column = lane&15.
  int lh = lane >> 4;
  int n0 = bn + wc * 32 + (lane & 15);
  float b0 = bias[n0];
  float b1 = bias[n0 + 16];
#pragma unroll
  for (int j = 0; j < 8; ++j) {
    int m0 = bm + wr * 32 + j + lh * 8;
    int m1 = m0 + 16;
    float v00 = acc00[j] + b0; v00 = v00 > 0.f ? v00 : 0.f;
    float v01 = acc01[j] + b1; v01 = v01 > 0.f ? v01 : 0.f;
    float v10 = acc10[j] + b0; v10 = v10 > 0.f ? v10 : 0.f;
    float v11 = acc11[j] + b1; v11 = v11 > 0.f ? v11 : 0.f;
    if (C16) {
      C16[(size_t)m0 * N + n0]      = (_Float16)v00;
      C16[(size_t)m0 * N + n0 + 16] = (_Float16)v01;
      C16[(size_t)m1 * N + n0]      = (_Float16)v10;
      C16[(size_t)m1 * N + n0 + 16] = (_Float16)v11;
    }
    if (C32) {
      C32[(size_t)m0 * N + n0]      = v00;
      C32[(size_t)m0 * N + n0 + 16] = v01;
      C32[(size_t)m1 * N + n0]      = v10;
      C32[(size_t)m1 * N + n0 + 16] = v11;
    }
  }
}

// ---------------------------------------------------------------------------
// Flash attention: one wave per (b, h, 16-query tile).
// ---------------------------------------------------------------------------
__global__ __launch_bounds__(32) void mha_attention(
    const _Float16* __restrict__ Q, const _Float16* __restrict__ Km,
    const _Float16* __restrict__ V, _Float16* __restrict__ O) {
  __shared__ _Float16 Pt[16 * 32];    // P tile, cols 16..31 stay zero
  __shared__ _Float16 Vt[128 * 16];   // V tile transposed: [dim][key]

  int lane = threadIdx.x;
  int qt = blockIdx.x, h = blockIdx.y, b = blockIdx.z;
  size_t baseQ  = ((size_t)(b * SEQ + qt * 16)) * ADIM + h * HDIM;
  size_t baseKV = ((size_t)(b * SEQ)) * ADIM + h * HDIM;

  for (int i = lane; i < 16 * 16; i += 32)
    Pt[(i >> 4) * 32 + 16 + (i & 15)] = (_Float16)0.f;
  __syncthreads();

  v16h qf[4];
#pragma unroll
  for (int t = 0; t < 4; ++t) qf[t] = frag_ld(Q + baseQ + t * 32, ADIM);

  float mrow[8], lrow[8];
  v8f o[8];
  v8f zf = {};
#pragma unroll
  for (int j = 0; j < 8; ++j) { mrow[j] = -1e30f; lrow[j] = 0.f; o[j] = zf; }

  int r  = lane & 15;
  int ko = (lane < 16) ? 0 : 8;
  int lh = lane >> 4;

  for (int kt = 0; kt < SEQ / 16; ++kt) {
    // ---- scores = Q . K^T ----
    v8f s = zf;
#pragma unroll
    for (int t = 0; t < 4; ++t) {
      const _Float16* kp =
          Km + baseKV + (size_t)(kt * 16 + r) * ADIM + t * 32 + ko;
      v8h lo = *(const v8h*)kp;
      v8h hi = *(const v8h*)(kp + 16);
      v16h bf = __builtin_shufflevector(lo, hi, 0,1,2,3,4,5,6,7,
                                        8,9,10,11,12,13,14,15);
      s = __builtin_amdgcn_wmma_f32_16x16x32_f16(false, qf[t], false, bf,
                                                 (short)0, s, false, false);
    }

    // ---- stage V tile transposed into LDS ----
#pragma unroll
    for (int it = 0; it < 8; ++it) {
      int id  = it * 32 + lane;
      int key = id >> 4;
      int ch  = id & 15;
      v8h vv = *(const v8h*)(V + baseKV + (size_t)(kt * 16 + key) * ADIM + ch * 8);
#pragma unroll
      for (int e = 0; e < 8; ++e) Vt[(ch * 8 + e) * 16 + key] = vv[e];
    }

    // ---- streaming softmax (row stats via 16-lane shuffle reductions) ----
    float alpha[8];
#pragma unroll
    for (int j = 0; j < 8; ++j) {
      float sv = s[j] * INV_SQRT_D;
      float mx = sv;
      mx = fmaxf(mx, __shfl_xor(mx, 1, 32));
      mx = fmaxf(mx, __shfl_xor(mx, 2, 32));
      mx = fmaxf(mx, __shfl_xor(mx, 4, 32));
      mx = fmaxf(mx, __shfl_xor(mx, 8, 32));
      float mnew = fmaxf(mrow[j], mx);
      float p = __expf(sv - mnew);
      float ps = p;
      ps += __shfl_xor(ps, 1, 32);
      ps += __shfl_xor(ps, 2, 32);
      ps += __shfl_xor(ps, 4, 32);
      ps += __shfl_xor(ps, 8, 32);
      alpha[j] = __expf(mrow[j] - mnew);
      lrow[j]  = lrow[j] * alpha[j] + ps;
      mrow[j]  = mnew;
      Pt[(j + lh * 8) * 32 + (lane & 15)] = (_Float16)p;
    }
    __syncthreads();

#pragma unroll
    for (int c = 0; c < 8; ++c)
#pragma unroll
      for (int j = 0; j < 8; ++j) o[c][j] *= alpha[j];

    // ---- O += P . V ----
    v16h pf = frag_ld(Pt, 32);   // upper K-half is zero
    v8h zh = {};
#pragma unroll
    for (int c = 0; c < 8; ++c) {
      v8h lo = *(const v8h*)(&Vt[(c * 16 + r) * 16 + ko]);
      v16h bf = __builtin_shufflevector(lo, zh, 0,1,2,3,4,5,6,7,
                                        8,9,10,11,12,13,14,15);
      o[c] = __builtin_amdgcn_wmma_f32_16x16x32_f16(false, pf, false, bf,
                                                    (short)0, o[c], false, false);
    }
    __syncthreads();
  }

#pragma unroll
  for (int j = 0; j < 8; ++j) lrow[j] = 1.f / lrow[j];
#pragma unroll
  for (int c = 0; c < 8; ++c)
#pragma unroll
    for (int j = 0; j < 8; ++j) {
      int m = qt * 16 + j + lh * 8;
      int d = h * HDIM + c * 16 + (lane & 15);
      O[((size_t)(b * SEQ + m)) * ADIM + d] = (_Float16)(o[c][j] * lrow[j]);
    }
}

// ---------------------------------------------------------------------------
extern "C" void kernel_launch(void* const* d_in, const int* in_sizes, int n_in,
                              void* d_out, int out_size, void* d_ws,
                              size_t ws_size, hipStream_t stream) {
  (void)in_sizes; (void)n_in; (void)out_size; (void)ws_size;
  const float* x   = (const float*)d_in[0];
  const float* Wq0 = (const float*)d_in[1];
  const float* bq0 = (const float*)d_in[2];
  const float* Wk0 = (const float*)d_in[3];
  const float* bk0 = (const float*)d_in[4];
  const float* Wv0 = (const float*)d_in[5];
  const float* bv0 = (const float*)d_in[6];
  const float* Wq  = (const float*)d_in[7];
  const float* bq  = (const float*)d_in[8];
  const float* Wk  = (const float*)d_in[9];
  const float* bk  = (const float*)d_in[10];
  const float* Wv  = (const float*)d_in[11];
  const float* bv  = (const float*)d_in[12];
  const float* Wo  = (const float*)d_in[13];
  const float* bo  = (const float*)d_in[14];
  float* out = (float*)d_out;

  const size_t MT = (size_t)BATCH * SEQ;   // 8192 rows

  char* ws = (char*)d_ws;
  size_t off = 0;
  auto alloc = [&](size_t halfs) {
    _Float16* p = (_Float16*)(ws + off);
    off += (halfs * 2 + 255) & ~(size_t)255;
    return p;
  };
  _Float16* X16  = alloc(MT * E0);
  _Float16* Q16  = alloc(MT * ADIM);
  _Float16* K16  = alloc(MT * ADIM);
  _Float16* V16  = alloc(MT * ADIM);
  _Float16* A16  = alloc(MT * ADIM);
  _Float16* Y16  = alloc(MT * ADIM);
  _Float16* Wq0h = alloc((size_t)E0 * ADIM);
  _Float16* Wk0h = alloc((size_t)E0 * ADIM);
  _Float16* Wv0h = alloc((size_t)E0 * ADIM);
  _Float16* Wqh  = alloc((size_t)2 * ADIM * ADIM);
  _Float16* Wkh  = alloc((size_t)2 * ADIM * ADIM);
  _Float16* Wvh  = alloc((size_t)2 * ADIM * ADIM);
  _Float16* Woh  = alloc((size_t)3 * ADIM * ADIM);

  auto cvt = [&](const float* s, _Float16* d, size_t n) {
    int blocks = (int)((n / 4 + 255) / 256);
    if (blocks > 4096) blocks = 4096;
    cvt_f32_to_f16<<<blocks, 256, 0, stream>>>(s, d, (long long)n);
  };
  cvt(x,   X16,  MT * E0);
  cvt(Wq0, Wq0h, (size_t)E0 * ADIM);
  cvt(Wk0, Wk0h, (size_t)E0 * ADIM);
  cvt(Wv0, Wv0h, (size_t)E0 * ADIM);
  cvt(Wq,  Wqh,  (size_t)2 * ADIM * ADIM);
  cvt(Wk,  Wkh,  (size_t)2 * ADIM * ADIM);
  cvt(Wv,  Wvh,  (size_t)2 * ADIM * ADIM);
  cvt(Wo,  Woh,  (size_t)3 * ADIM * ADIM);

  auto gemm = [&](const _Float16* A, const _Float16* W, const float* bias,
                  _Float16* C16, float* C32, int M, int N, int K) {
    dim3 grid(N / 64, M / 128);
    gemm_relu<<<grid, 256, 0, stream>>>(A, W, bias, C16, C32, M, N, K);
  };
  dim3 agrid(SEQ / 16, HEADS, BATCH);

  // ---- layer 0 (K = 256) ----
  gemm(X16, Wq0h, bq0, Q16, nullptr, (int)MT, ADIM, E0);
  gemm(X16, Wk0h, bk0, K16, nullptr, (int)MT, ADIM, E0);
  gemm(X16, Wv0h, bv0, V16, nullptr, (int)MT, ADIM, E0);
  mha_attention<<<agrid, 32, 0, stream>>>(Q16, K16, V16, A16);
  gemm(A16, Woh, bo, Y16, nullptr, (int)MT, ADIM, ADIM);

  // ---- layers 1..2 (K = 1024) ----
  for (int i = 0; i < 2; ++i) {
    size_t wofs = (size_t)i * ADIM * ADIM;
    gemm(Y16, Wqh + wofs, bq + i * ADIM, Q16, nullptr, (int)MT, ADIM, ADIM);
    gemm(Y16, Wkh + wofs, bk + i * ADIM, K16, nullptr, (int)MT, ADIM, ADIM);
    gemm(Y16, Wvh + wofs, bv + i * ADIM, V16, nullptr, (int)MT, ADIM, ADIM);
    mha_attention<<<agrid, 32, 0, stream>>>(Q16, K16, V16, A16);
    bool last = (i == 1);
    gemm(A16, Woh + (size_t)(i + 1) * ADIM * ADIM, bo + (i + 1) * ADIM,
         last ? nullptr : Y16, last ? out : nullptr, (int)MT, ADIM, ADIM);
  }
}
